// SwarmFormerLayer_36928128811225
// MI455X (gfx1250) — matrix operand order repaired and checked by reading, not compiled, not run on hardware
//
#include <hip/hip_runtime.h>
#include <cmath>

#define DEV __device__ __forceinline__

typedef __attribute__((ext_vector_type(16))) __bf16 v16bf;
typedef __attribute__((ext_vector_type(8)))  float  v8f;

namespace {

constexpr int D    = 64;
constexpr int NB   = 64;     // batch
constexpr int L    = 8192;   // sequence
constexpr int NC   = 1024;   // clusters per batch (L/8)
constexpr int CS   = 8;      // cluster size
constexpr int TILE = 128;    // tokens per block

// packed-weight unit ids (each unit = one 64x64 matrix in B-fragment layout)
constexpr int U_W1 = 0, U_W2 = 1, U_G1LO = 2, U_G1HI = 3, U_G2 = 4;
constexpr int U_QW = 5, U_KW = 6, U_VW = 7, U_BP = 8;
constexpr int U_B1LO = 9, U_B1HI = 10, U_B2 = 11;
constexpr int NUNIT = 12;
constexpr int UELEMS = 8 * 32 * 16;   // (s,t) tiles x lanes x 16 bf16 = 4096

DEV v8f vzero() {
  v8f z;
#pragma unroll
  for (int i = 0; i < 8; ++i) z[i] = 0.f;
  return z;
}

DEV void acc4zero(v8f a[4]) {
#pragma unroll
  for (int t = 0; t < 4; ++t) a[t] = vzero();
}

DEV void wmma_bf16(const v16bf& a, const v16bf& b, v8f& c) {
  // D = A(16x32 bf16) * B(32x16 bf16) + C(16x16 f32)
  c = __builtin_amdgcn_wmma_f32_16x16x32_bf16(false, a, false, b, (short)0, c,
                                              false, false);
}

// --- A fragment (16x32, bf16). Lane l: row = l%16; low-half VGPRs hold
//     K = (l>=16?16:0)+0..7, high-half hold K = (l>=16?16:0)+8+0..7. ---
template <class F>
DEV v16bf load_a(F&& src, int kofs) {
  const int lane = threadIdx.x & 31;
  const int row  = lane & 15;
  const int b0   = (lane >> 4) << 4;
  v16bf a;
#pragma unroll
  for (int e = 0; e < 8; ++e) {
    a[e]     = (__bf16)src(row, kofs + b0 + e);
    a[e + 8] = (__bf16)src(row, kofs + b0 + 8 + e);
  }
  return a;
}

template <class F>
DEV void load_a2(F&& src, v16bf a[2]) {
  a[0] = load_a(src, 0);
  a[1] = load_a(src, 32);
}

// --- B fragment (32x16, bf16) built from arbitrary source (runtime data). ---
template <class F>
DEV v16bf load_b(F&& src, int kofs, int nofs) {
  const int lane = threadIdx.x & 31;
  const int col  = lane & 15;
  const int kb   = (lane >> 4) << 4;
  v16bf b;
#pragma unroll
  for (int e = 0; e < 16; ++e)
    b[e] = (__bf16)src(kofs + kb + e, nofs + col);
  return b;
}

// 16x64 @ 64x64 GEMM accumulate from pre-packed bf16 fragments (one 32-byte
// contiguous vector load per lane per fragment; stays resident in WGP$/L2).
DEV void gemm_p(const v16bf a[2], const __bf16* __restrict__ frag, v8f acc[4]) {
  const int lane = threadIdx.x & 31;
#pragma unroll
  for (int s = 0; s < 2; ++s) {
#pragma unroll
    for (int t = 0; t < 4; ++t) {
      v16bf b = *(const v16bf*)(frag + (((s << 2) | t) * 32 + lane) * 16);
      wmma_bf16(a[s], b, acc[t]);
    }
  }
}

DEV float gelu_f(float x) { return 0.5f * x * (1.f + erff(x * 0.7071067811865475f)); }
DEV float sigm_f(float x) { return __builtin_amdgcn_rcpf(1.f + __expf(-x)); }

// Write 16x64 C tiles (+bias, optional exact GELU) to row-major [m*D+n].
DEV void store_rows(float* dst, const v8f acc[4], const float* __restrict__ bias,
                    int act_gelu) {
  const int lane = threadIdx.x & 31;
  const int n0 = lane & 15, mh = (lane >> 4) << 3;
#pragma unroll
  for (int t = 0; t < 4; ++t) {
    const int n = n0 + 16 * t;
    const float bv = bias[n];
#pragma unroll
    for (int v = 0; v < 8; ++v) {
      float r = acc[t][v] + bv;
      dst[(v + mh) * D + n] = act_gelu ? gelu_f(r) : r;
    }
  }
}

// Write 16x64 C tiles transposed: dst[n*ld + m] (+bias).
DEV void store_cols(float* dst, int ld, const v8f acc[4],
                    const float* __restrict__ bias) {
  const int lane = threadIdx.x & 31;
  const int n0 = lane & 15, mh = (lane >> 4) << 3;
#pragma unroll
  for (int t = 0; t < 4; ++t) {
    const int n = n0 + 16 * t;
    const float bv = bias[n];
#pragma unroll
    for (int v = 0; v < 8; ++v) dst[(size_t)n * ld + (v + mh)] = acc[t][v] + bv;
  }
}

// One LocalSwarmAggregator micro-iteration for a 16-row chunk.
// xsrc(m,k): x at chunk-relative row m (valid m in [-1,16]); dst(m,n,v): writer.
template <class XS, class DS>
DEV void swarm_chunk(XS&& xsrc, DS&& dst, float* tmp, float* upd,
                     const __bf16* __restrict__ wp,
                     const float* b1, const float* b2,
                     const float* g1b, const float* g2b) {
  v16bf a[2];
  v8f acc[4];

  // h1 = gelu(nbr @ w1 + b1)
  load_a2([&](int m, int k) {
            return (xsrc(m - 1, k) + xsrc(m, k) + xsrc(m + 1, k)) * (1.f / 3.f);
          }, a);
  acc4zero(acc);
  gemm_p(a, wp + U_W1 * UELEMS, acc);
  store_rows(tmp, acc, b1, 1);

  // update = h1 @ w2 + b2
  load_a2([&](int m, int k) { return tmp[m * D + k]; }, a);
  acc4zero(acc);
  gemm_p(a, wp + U_W2 * UELEMS, acc);
  store_rows(upd, acc, b2, 0);

  // g = gelu(x @ g1w[0:64] + update @ g1w[64:128] + g1b)
  load_a2([&](int m, int k) { return xsrc(m, k); }, a);
  acc4zero(acc);
  gemm_p(a, wp + U_G1LO * UELEMS, acc);
  load_a2([&](int m, int k) { return upd[m * D + k]; }, a);
  gemm_p(a, wp + U_G1HI * UELEMS, acc);
  store_rows(tmp, acc, g1b, 1);

  // gates = sigmoid(g @ g2w + g2b); xnew = x + gates*(update - x)
  load_a2([&](int m, int k) { return tmp[m * D + k]; }, a);
  acc4zero(acc);
  gemm_p(a, wp + U_G2 * UELEMS, acc);

  const int lane = threadIdx.x & 31;
  const int n0 = lane & 15, mh = (lane >> 4) << 3;
#pragma unroll
  for (int t = 0; t < 4; ++t) {
    const int n = n0 + 16 * t;
    const float bg = g2b[n];
#pragma unroll
    for (int v = 0; v < 8; ++v) {
      const int m = v + mh;
      float g = sigm_f(acc[t][v] + bg);
      float xv = xsrc(m, n);
      float uv = upd[m * D + n];
      dst(m, n, xv + g * (uv - xv));
    }
  }
}

// ---------------- Kernel 0: pack weights into bf16 B-fragment layout -------
struct PackArgs { const float* w[NUNIT]; };

__global__ __launch_bounds__(256) void k_pack(PackArgs pa, __bf16* __restrict__ wp) {
  const int u = blockIdx.x;
  const float* __restrict__ W = pa.w[u];
  for (int idx = threadIdx.x; idx < UELEMS; idx += 256) {
    const int st   = idx >> 9;          // (s*4+t)
    const int lane = (idx >> 4) & 31;
    const int e    = idx & 15;
    const int s = st >> 2, t = st & 3;
    const int k = 32 * s + ((lane >> 4) << 4) + e;
    const int n = 16 * t + (lane & 15);
    wp[u * UELEMS + idx] = (__bf16)W[k * D + n];
  }
}

// ---------------- Kernel 1: fused 2x local aggregator + cluster means ------
__global__ __launch_bounds__(256) void k_local(
    const float* __restrict__ x, const __bf16* __restrict__ wp,
    const float* __restrict__ b1, const float* __restrict__ b2,
    const float* __restrict__ g1b, const float* __restrict__ g2b,
    float* __restrict__ out, float* __restrict__ reps) {
  __shared__ float xbA[(TILE + 32) * D];   // rows -16 .. TILE+15
  __shared__ float xbB[(TILE + 16) * D];   // rows  -8 .. TILE+7
  __shared__ float scrT[8][16 * D];
  __shared__ float scrU[8][16 * D];

  const int blk  = blockIdx.x;
  const int bz   = blk >> 6;            // L/TILE = 64 tiles per batch
  const int tb   = (blk & 63) * TILE;   // token base within batch
  const int tid  = threadIdx.x;
  const int wave = tid >> 5;
  const size_t xbase = (size_t)bz * L * D;

  // stage tile + halo (roll wraps within the sequence)
  for (int idx = tid; idx < (TILE + 32) * D; idx += 256) {
    const int r = idx >> 6, c = idx & 63;
    const int g = (tb + r - 16 + L) & (L - 1);
    xbA[idx] = x[xbase + (size_t)g * D + c];
  }
  __syncthreads();

  float* tmp = scrT[wave];
  float* upd = scrU[wave];
  float* xA = xbA + 16 * D;   // row r at xA[r*D]
  float* xB = xbB + 8 * D;

  // iteration 1: rows [-8, TILE+8) -> xbB  (9 chunks over 8 waves)
  for (int c8 = wave; c8 < 9; c8 += 8) {
    const int r0 = -8 + 16 * c8;
    swarm_chunk(
        [&](int m, int k) { return xA[(r0 + m) * D + k]; },
        [&](int m, int n, float v) { xB[(r0 + m) * D + n] = v; },
        tmp, upd, wp, b1, b2, g1b, g2b);
  }
  __syncthreads();

  // iteration 2: rows [0, TILE) -> xbA
  {
    const int r0 = wave * 16;
    swarm_chunk(
        [&](int m, int k) { return xB[(r0 + m) * D + k]; },
        [&](int m, int n, float v) { xA[(r0 + m) * D + n] = v; },
        tmp, upd, wp, b1, b2, g1b, g2b);
  }
  __syncthreads();

  // write x_after and cluster means
  for (int idx = tid; idx < TILE * D; idx += 256) {
    const int r = idx >> 6, c = idx & 63;
    out[xbase + (size_t)(tb + r) * D + c] = xA[r * D + c];
  }
  for (int idx = tid; idx < (TILE / CS) * D; idx += 256) {
    const int cl = idx >> 6, c = idx & 63;
    float s = 0.f;
#pragma unroll
    for (int j = 0; j < CS; ++j) s += xA[(cl * CS + j) * D + c];
    reps[((size_t)bz * NC + tb / CS + cl) * D + c] = s * 0.125f;
  }
}

// ---------------- Kernel 2: Q/K^T/V projections ----------------------------
__global__ __launch_bounds__(256) void k_qkv(
    const float* __restrict__ reps, const __bf16* __restrict__ wp,
    const float* __restrict__ qb, const float* __restrict__ kbias,
    const float* __restrict__ vb,
    float* __restrict__ Q, float* __restrict__ KT, float* __restrict__ Vp) {
  const size_t row0 = (size_t)blockIdx.x * TILE + (threadIdx.x >> 5) * 16;
  const int bz   = (int)(row0 >> 10);       // cluster row -> batch
  const int key0 = (int)(row0 & (NC - 1));
  v16bf a[2];
  load_a2([&](int m, int k) { return reps[(row0 + m) * D + k]; }, a);
  v8f acc[4];
  acc4zero(acc); gemm_p(a, wp + U_QW * UELEMS, acc);
  store_rows(Q + row0 * D, acc, qb, 0);
  acc4zero(acc); gemm_p(a, wp + U_KW * UELEMS, acc);
  store_cols(KT + (size_t)bz * D * NC + key0, NC, acc, kbias);   // KT[b][d][key]
  acc4zero(acc); gemm_p(a, wp + U_VW * UELEMS, acc);
  store_rows(Vp + row0 * D, acc, vb, 0);
}

// ---------------- Kernel 3: flash cluster attention + bp projection --------
__global__ __launch_bounds__(256) void k_attn(
    const float* __restrict__ Q, const float* __restrict__ KT,
    const float* __restrict__ Vp, const __bf16* __restrict__ wp,
    const float* __restrict__ bpb, float* __restrict__ cu) {
  __shared__ float pscr[8][16 * 32];
  __shared__ float uscr[8][16 * D];

  const int bz   = blockIdx.x >> 3;    // NC/TILE = 8 q-tiles per batch
  const int qt   = blockIdx.x & 7;
  const int wave = threadIdx.x >> 5;
  const int lane = threadIdx.x & 31;
  const size_t qrow   = (size_t)bz * NC + qt * TILE + wave * 16;
  const size_t kvbase = (size_t)bz * NC;
  const size_t ktbase = (size_t)bz * D * NC;

  v16bf aq[2];
  load_a2([&](int m, int k) { return Q[(qrow + m) * D + k]; }, aq);

  float mrow[8], lrow[8], alpha[8];
#pragma unroll
  for (int v = 0; v < 8; ++v) { mrow[v] = -3.0e38f; lrow[v] = 0.f; }
  v8f accO[4];
  acc4zero(accO);

  float* ps = pscr[wave];
  const int n0 = lane & 15, mh = (lane >> 4) << 3;

  for (int kb0 = 0; kb0 < NC; kb0 += 32) {
    if (kb0 + 32 < NC) {   // stream-ahead hints for next K/V block
      __builtin_prefetch(&KT[ktbase + (size_t)(lane & 15) * NC + kb0 + 32], 0, 1);
      __builtin_prefetch(&Vp[(kvbase + kb0 + 32 + lane) * D], 0, 1);
    }
    // scores for 32 keys: two 16x16 tiles; KT gives coalesced per-lane reads
    v8f s0 = vzero(), s1 = vzero();
#pragma unroll
    for (int s = 0; s < 2; ++s) {
      v16bf b0 = load_b([&](int k, int n) {
                          return KT[ktbase + (size_t)k * NC + kb0 + n];
                        }, 32 * s, 0);
      wmma_bf16(aq[s], b0, s0);
      v16bf b1 = load_b([&](int k, int n) {
                          return KT[ktbase + (size_t)k * NC + kb0 + 16 + n];
                        }, 32 * s, 0);
      wmma_bf16(aq[s], b1, s1);
    }
    // online softmax (row M lives in element index; N across 16-lane halves)
#pragma unroll
    for (int v = 0; v < 8; ++v) {
      float x0 = s0[v] * 0.125f, x1 = s1[v] * 0.125f;
      float mv = fmaxf(x0, x1);
#pragma unroll
      for (int o = 1; o < 16; o <<= 1) mv = fmaxf(mv, __shfl_xor(mv, o, 32));
      float mn = fmaxf(mrow[v], mv);
      alpha[v] = __expf(mrow[v] - mn);
      float p0 = __expf(x0 - mn), p1 = __expf(x1 - mn);
      float rs = p0 + p1;
#pragma unroll
      for (int o = 1; o < 16; o <<= 1) rs += __shfl_xor(rs, o, 32);
      lrow[v] = lrow[v] * alpha[v] + rs;
      mrow[v] = mn;
      const int m = v + mh;
      ps[m * 32 + n0]      = p0;
      ps[m * 32 + n0 + 16] = p1;
    }
#pragma unroll
    for (int t = 0; t < 4; ++t)
#pragma unroll
      for (int v = 0; v < 8; ++v) accO[t][v] *= alpha[v];

    // accO += P(16x32) @ V(32x64)
    v16bf ap = load_a([&](int m, int k) { return ps[m * 32 + k]; }, 0);
#pragma unroll
    for (int t = 0; t < 4; ++t) {
      v16bf bv = load_b([&](int k, int n) {
                          return Vp[(kvbase + kb0 + k) * D + n];
                        }, 0, 16 * t);
      wmma_bf16(ap, bv, accO[t]);
    }
  }

  // normalize -> uscr, then cu = updated @ bp_w + bp_b
  float* us = uscr[wave];
#pragma unroll
  for (int t = 0; t < 4; ++t) {
#pragma unroll
    for (int v = 0; v < 8; ++v) {
      float inv = __builtin_amdgcn_rcpf(lrow[v]);
      us[(v + mh) * D + n0 + 16 * t] = accO[t][v] * inv;
    }
  }

  v16bf au[2];
  load_a2([&](int m, int k) { return us[m * D + k]; }, au);
  v8f acc[4];
  acc4zero(acc);
  gemm_p(au, wp + U_BP * UELEMS, acc);
  store_rows(cu + qrow * D, acc, bpb, 0);
}

// ---------------- Kernel 4: broadcast gate + final blend (in place) --------
__global__ __launch_bounds__(256) void k_bcast(
    const float* __restrict__ cu, const __bf16* __restrict__ wp,
    const float* __restrict__ g1b, const float* __restrict__ g2b,
    float* __restrict__ out) {
  __shared__ float scr[8][16 * D];

  const int blk  = blockIdx.x;
  const int bz   = blk >> 6;
  const int tb   = (blk & 63) * TILE;
  const int wave = threadIdx.x >> 5;
  const int lane = threadIdx.x & 31;
  const int r0   = tb + wave * 16;                 // token within batch
  const size_t grow  = (size_t)bz * L + r0;        // global token row
  const size_t crow0 = (size_t)bz * NC;

  v16bf ax[2], ac[2];
  load_a2([&](int m, int k) { return out[(grow + m) * D + k]; }, ax);
  load_a2([&](int m, int k) {
            return cu[(crow0 + ((r0 + m) >> 3)) * D + k];
          }, ac);

  v8f acc[4];
  acc4zero(acc);
  gemm_p(ax, wp + U_B1LO * UELEMS, acc);
  gemm_p(ac, wp + U_B1HI * UELEMS, acc);
  store_rows(scr[wave], acc, g1b, 1);

  v16bf ag[2];
  load_a2([&](int m, int k) { return scr[wave][m * D + k]; }, ag);
  acc4zero(acc);
  gemm_p(ag, wp + U_B2 * UELEMS, acc);

  const int n0 = lane & 15, mh = (lane >> 4) << 3;
#pragma unroll
  for (int t = 0; t < 4; ++t) {
    const int n = n0 + 16 * t;
    const float bg = g2b[n];
#pragma unroll
    for (int v = 0; v < 8; ++v) {
      const int m = v + mh;
      float g = sigm_f(acc[t][v] + bg);
      float xv = out[(grow + m) * D + n];
      float cv = cu[(crow0 + ((r0 + m) >> 3)) * D + n];
      out[(grow + m) * D + n] = xv + g * (cv - xv);
    }
  }
}

}  // namespace

extern "C" void kernel_launch(void* const* d_in, const int* in_sizes, int n_in,
                              void* d_out, int out_size, void* d_ws, size_t ws_size,
                              hipStream_t stream) {
  (void)in_sizes; (void)n_in; (void)out_size; (void)ws_size;
  const float* x      = (const float*)d_in[0];
  const float* la_w1  = (const float*)d_in[1];
  const float* la_b1  = (const float*)d_in[2];
  const float* la_w2  = (const float*)d_in[3];
  const float* la_b2  = (const float*)d_in[4];
  const float* la_g1w = (const float*)d_in[5];
  const float* la_g1b = (const float*)d_in[6];
  const float* la_g2w = (const float*)d_in[7];
  const float* la_g2b = (const float*)d_in[8];
  const float* qw   = (const float*)d_in[9];
  const float* qb   = (const float*)d_in[10];
  const float* kw   = (const float*)d_in[11];
  const float* kb   = (const float*)d_in[12];
  const float* vw   = (const float*)d_in[13];
  const float* vb   = (const float*)d_in[14];
  const float* bp_w = (const float*)d_in[15];
  const float* bp_b = (const float*)d_in[16];
  const float* bg1w = (const float*)d_in[17];
  const float* bg1b = (const float*)d_in[18];
  const float* bg2w = (const float*)d_in[19];
  const float* bg2b = (const float*)d_in[20];

  float* out = (float*)d_out;

  __bf16* wp = (__bf16*)d_ws;                       // 12 * 4096 bf16 = 96 KB
  float* fws = (float*)((char*)d_ws + 131072);      // 128 KB aligned region
  const size_t repsN = (size_t)NB * NC * D;         // 4M floats each
  float* reps = fws;
  float* Q    = reps + repsN;
  float* KT   = Q + repsN;
  float* Vp   = KT + repsN;
  float* cu   = Vp + repsN;

  PackArgs pa;
  pa.w[U_W1] = la_w1;  pa.w[U_W2] = la_w2;
  pa.w[U_G1LO] = la_g1w; pa.w[U_G1HI] = la_g1w + D * D;
  pa.w[U_G2] = la_g2w;
  pa.w[U_QW] = qw; pa.w[U_KW] = kw; pa.w[U_VW] = vw; pa.w[U_BP] = bp_w;
  pa.w[U_B1LO] = bg1w; pa.w[U_B1HI] = bg1w + D * D; pa.w[U_B2] = bg2w;

  k_pack<<<NUNIT, 256, 0, stream>>>(pa, wp);
  k_local<<<NB * (L / TILE), 256, 0, stream>>>(
      x, wp, la_b1, la_b2, la_g1b, la_g2b, out, reps);
  k_qkv<<<(NB * NC) / TILE, 256, 0, stream>>>(
      reps, wp, qb, kb, vb, Q, KT, Vp);
  k_attn<<<NB * (NC / TILE), 256, 0, stream>>>(
      Q, KT, Vp, wp, bp_b, cu);
  k_bcast<<<NB * (L / TILE), 256, 0, stream>>>(
      cu, wp, bg1b, bg2b, out);
}
